// TransformerBlock_10771777978879
// MI455X (gfx1250) — matrix operand order repaired and checked by reading, not compiled, not run on hardware
//
#include <hip/hip_runtime.h>

// ---------------------------------------------------------------------------
// Types for CDNA5 WMMA (wave32): v_wmma_f32_16x16x32_bf16
// ---------------------------------------------------------------------------
typedef __bf16 bf16;
typedef bf16  v8bf  __attribute__((ext_vector_type(8)));
typedef bf16  v16bf __attribute__((ext_vector_type(16)));
typedef float v8f   __attribute__((ext_vector_type(8)));

__device__ __forceinline__ v8f wmma_bf16(v16bf a, v16bf b, v8f c) {
    // 8 args: (neg_a, A, neg_b, B, c_mod, C, reuse_a, reuse_b)
    return __builtin_amdgcn_wmma_f32_16x16x32_bf16(false, a, false, b, (short)0, c,
                                                   false, false);
}

union V16U { v16bf v; v8bf h[2]; };

// A fragment (16x32, 16-bit): lane l holds row M = l%16; elems 0..15 =
// K g*16..+15 (g = l/16). K-runs contiguous -> two b128 loads.
__device__ __forceinline__ v16bf load_a_frag(const bf16* base, int row, int stride, int g) {
    V16U u;
    u.h[0] = *(const v8bf*)(base + (size_t)row * stride + g * 16);
    u.h[1] = *(const v8bf*)(base + (size_t)row * stride + g * 16 + 8);
    return u.v;
}

// B fragment (32x16, 16-bit): lane l holds column N = l%16; elems 0..15 =
// K (l/16)*16..+15 contiguous. `colbase` = start of that column's K-run.
__device__ __forceinline__ v16bf load_b_frag(const bf16* colbase, int g) {
    V16U u;
    u.h[0] = *(const v8bf*)(colbase + g * 16);
    u.h[1] = *(const v8bf*)(colbase + g * 16 + 8);
    return u.v;
}

// CDNA5 async global->LDS copy (ASYNCcnt-tracked, no VGPR staging).
// INST_OFFSET is added to BOTH the LDS and global addresses (ISA 08 §4.4).
__device__ __forceinline__ void async_g2l_b128(unsigned lds_off, const void* gaddr) {
    asm volatile("global_load_async_to_lds_b128 %0, %1, off" ::"v"(lds_off), "v"(gaddr)
                 : "memory");
}
__device__ __forceinline__ void async_g2l_b128_o16(unsigned lds_off, const void* gaddr) {
    asm volatile("global_load_async_to_lds_b128 %0, %1, off offset:16" ::"v"(lds_off),
                     "v"(gaddr)
                 : "memory");
}
__device__ __forceinline__ void wait_async0() {
    asm volatile("s_wait_asynccnt 0x0" ::: "memory");
}

// ---------------------------------------------------------------------------
// Transpose + convert: in fp32 [K][N] -> out bf16 [N][K] (32x32 LDS tiles).
// One-time weight prep so GEMM B-staging is a contiguous (async-able) copy.
// ---------------------------------------------------------------------------
__global__ __launch_bounds__(256) void cvt_t_kernel(const float* __restrict__ in,
                                                    bf16* __restrict__ out, int K, int N) {
    __shared__ float t[32][33];
    int k0 = blockIdx.x * 32, n0 = blockIdx.y * 32;
    int tx = threadIdx.x & 31, ty = threadIdx.x >> 5;  // 32 x 8
#pragma unroll
    for (int i = 0; i < 32; i += 8) t[ty + i][tx] = in[(size_t)(k0 + ty + i) * N + n0 + tx];
    __syncthreads();
#pragma unroll
    for (int i = 0; i < 32; i += 8)
        out[(size_t)(n0 + ty + i) * K + k0 + tx] = (bf16)t[tx][ty + i];
}

// ---------------------------------------------------------------------------
// LayerNorm -> bf16 rows. One 256-thread block per row, D = 1024.
// ---------------------------------------------------------------------------
__global__ __launch_bounds__(256) void ln_bf16_kernel(const float* __restrict__ X,
                                                      const float* __restrict__ gam,
                                                      const float* __restrict__ bet,
                                                      bf16* __restrict__ out, int D) {
    __shared__ float rs[256], rss[256];
    int row = blockIdx.x;
    const float* x = X + (size_t)row * D;
    float s = 0.f, ss = 0.f;
    for (int i = threadIdx.x; i < D; i += 256) {
        float v = x[i];
        s += v;
        ss += v * v;
    }
    rs[threadIdx.x] = s;
    rss[threadIdx.x] = ss;
    __syncthreads();
    for (int off = 128; off > 0; off >>= 1) {
        if (threadIdx.x < off) {
            rs[threadIdx.x]  += rs[threadIdx.x + off];
            rss[threadIdx.x] += rss[threadIdx.x + off];
        }
        __syncthreads();
    }
    float inv  = 1.0f / (float)D;
    float mu   = rs[0] * inv;
    float var  = rss[0] * inv - mu * mu;
    float rstd = rsqrtf(var + 1e-5f);
    for (int i = threadIdx.x; i < D; i += 256) {
        float v = (x[i] - mu) * rstd * gam[i] + bet[i];
        out[(size_t)row * D + i] = (bf16)v;
    }
}

// ---------------------------------------------------------------------------
// Tiled bf16 WMMA GEMM: C[M,N] = A[M,K] @ Wt[N,K]^T + bias (+res) (ReLU?)
// Block = 256 threads = 8 waves; block tile 128x128, K-step 32.
// Waves 4(M) x 2(N); each wave owns 32x64 (2x4 WMMA accumulators).
// Double-buffered LDS fed by global_load_async_to_lds_b128: tile t+1 streams
// while tile t is consumed by WMMA; one s_wait_asynccnt + barrier per step.
// ---------------------------------------------------------------------------
#define GM 128
#define GN 128
#define GK 32

__global__ __launch_bounds__(256) void gemm_bf16_kernel(
    const bf16* __restrict__ A, const bf16* __restrict__ Wt,  // Wt: [N][K]
    const float* __restrict__ bias, const float* __restrict__ res,
    float* __restrict__ outF, bf16* __restrict__ outB,
    int M, int N, int K, int relu) {
    __shared__ bf16 As[2][GM][GK];  // 16 KB
    __shared__ bf16 Bs[2][GN][GK];  // 16 KB

    const int tid  = threadIdx.x;
    const int wave = tid >> 5;
    const int lane = tid & 31;
    const int g    = lane >> 4;
    const int lr   = lane & 15;
    const int MB   = blockIdx.x * GM;
    const int NB   = blockIdx.y * GN;
    const int mW   = (wave & 3) * 32;
    const int nW   = (wave >> 2) * 64;

    // Staging assignment: thread t -> row = t/2, 16 elems (32 B) at (t&1)*16.
    const int srow = tid >> 1;
    const int shalf = (tid & 1) * 16;
    const bf16* gA = A + (size_t)(MB + srow) * K + shalf;
    const bf16* gB = Wt + (size_t)(NB + srow) * K + shalf;

    const int nt = K / GK;

    auto issue_tile = [&](int t, int buf) {
        unsigned la = (unsigned)(uintptr_t)&As[buf][srow][shalf];
        unsigned lb = (unsigned)(uintptr_t)&Bs[buf][srow][shalf];
        const bf16* pa = gA + (size_t)t * GK;
        const bf16* pb = gB + (size_t)t * GK;
        async_g2l_b128(la, pa);
        async_g2l_b128_o16(la, pa);
        async_g2l_b128(lb, pb);
        async_g2l_b128_o16(lb, pb);
    };

    v8f acc[2][4] = {};
    issue_tile(0, 0);

    for (int t = 0; t < nt; ++t) {
        int buf = t & 1;
        wait_async0();     // this thread's tile-t async copies have landed
        __syncthreads();   // all threads' copies landed; all done with buf^1
        if (t + 1 < nt) issue_tile(t + 1, buf ^ 1);  // overlaps with compute

        v16bf af[2], bfr[4];
#pragma unroll
        for (int i = 0; i < 2; ++i)
            af[i] = load_a_frag(&As[buf][0][0], mW + i * 16 + lr, GK, g);
#pragma unroll
        for (int j = 0; j < 4; ++j)
            bfr[j] = load_b_frag(&Bs[buf][nW + j * 16 + lr][0], g);
#pragma unroll
        for (int i = 0; i < 2; ++i)
#pragma unroll
            for (int j = 0; j < 4; ++j) acc[i][j] = wmma_bf16(af[i], bfr[j], acc[i][j]);
    }

    // Epilogue: C layout -> lane holds col = l%16, rows (l/16)*8 + r per VGPR r.
#pragma unroll
    for (int i = 0; i < 2; ++i) {
#pragma unroll
        for (int j = 0; j < 4; ++j) {
            int col = NB + nW + j * 16 + lr;
            float bv = bias ? bias[col] : 0.0f;
#pragma unroll
            for (int r = 0; r < 8; ++r) {
                int row = MB + mW + i * 16 + g * 8 + r;
                float v = acc[i][j][r] + bv;
                if (res)  v += res[(size_t)row * N + col];
                if (relu) v = fmaxf(v, 0.0f);
                if (outF) outF[(size_t)row * N + col] = v;
                if (outB) outB[(size_t)row * N + col] = (bf16)v;
            }
        }
    }
}

// ---------------------------------------------------------------------------
// Flash attention (causal), bf16 WMMA, wave32.
// qkv layout: [B*T][3072], cols 0..1023=Q (h*64+dk), 1024..2047=K, 2048..3071=V.
// Grid: (T/128, B*H). Block: 256 threads; wave w owns queries [qb+16w, +16).
// K tile staged with async global->LDS (contiguous rows == K^T B-frag order);
// V staged transposed via ds_store. Per-wave causal trip limit is wave-uniform
// so EXEC stays all-ones around WMMA.
// ---------------------------------------------------------------------------
__global__ __launch_bounds__(256) void attn_kernel(const bf16* __restrict__ qkv,
                                                   bf16* __restrict__ ctx,
                                                   int B, int T) {
    const int H = 16, DK = 64, QS = 3 * H * DK;  // row stride 3072
    const int bh = blockIdx.y;
    const int b  = bh / H;
    const int h  = bh % H;
    const int qb = blockIdx.x * 128;

    const int tid  = threadIdx.x;
    const int wave = tid >> 5;
    const int lane = tid & 31;
    const int g    = lane >> 4;
    const int lr   = lane & 15;
    const int q0   = qb + wave * 16;

    __shared__ bf16 Ks[32][64];      // key tile [key][dk]            (4 KB)
    __shared__ bf16 Vt[64][32];      // value tile transposed [dk][k] (4 KB)
    __shared__ bf16 Pst[8][16][32];  // per-wave P re-layout bounce   (8 KB)

    const bf16* qkv_b = qkv + (size_t)b * T * QS;
    const bf16* Qbase = qkv_b + h * DK;
    const bf16* Kbase = qkv_b + H * DK + h * DK;
    const bf16* Vbase = qkv_b + 2 * H * DK + h * DK;

    // Q fragments for both dk K-steps (A layout; per-lane K-runs contiguous).
    v16bf qf[2];
#pragma unroll
    for (int s = 0; s < 2; ++s) {
        const bf16* qrow = Qbase + (size_t)(q0 + lr) * QS + s * 32;
        V16U u;
        u.h[0] = *(const v8bf*)(qrow + g * 16);
        u.h[1] = *(const v8bf*)(qrow + g * 16 + 8);
        qf[s] = u.v;
    }

    v8f cacc[4] = {};        // ctx accumulators, 4 dk-chunks of 16
    float mrun[8], lrun[8];  // per-lane row stats (8 rows each, C layout)
#pragma unroll
    for (int r = 0; r < 8; ++r) { mrun[r] = -3.0e38f; lrun[r] = 0.0f; }

    const int kbmax   = (qb + 128) / 32;  // block-uniform trip count
    const int mylimit = (q0 + 15) / 32;   // wave-uniform causal limit

    // Staging assignment: thread -> key row tid/8, 8 dk elems (16 B) at (tid&7)*8.
    const int krow = tid >> 3;
    const int c0   = (tid & 7) * 8;
    const unsigned lk = (unsigned)(uintptr_t)&Ks[krow][c0];

    for (int kb = 0; kb < kbmax; ++kb) {
        __syncthreads();
        {
            int key = kb * 32 + krow;
            async_g2l_b128(lk, Kbase + (size_t)key * QS + c0);  // K: async copy
            v8bf vv = *(const v8bf*)(Vbase + (size_t)key * QS + c0);
#pragma unroll
            for (int j = 0; j < 8; ++j) Vt[c0 + j][krow] = vv[j];  // V: transpose
        }
        wait_async0();
        __syncthreads();

        if (kb <= mylimit) {
            // S = Q @ K^T : two 16(q)x16(key) tiles over the 32-key block.
            v8f s0 = {}, s1 = {};
#pragma unroll
            for (int s = 0; s < 2; ++s) {
                V16U u0, u1;
                u0.h[0] = *(const v8bf*)&Ks[lr][s * 32 + g * 16];
                u0.h[1] = *(const v8bf*)&Ks[lr][s * 32 + g * 16 + 8];
                u1.h[0] = *(const v8bf*)&Ks[16 + lr][s * 32 + g * 16];
                u1.h[1] = *(const v8bf*)&Ks[16 + lr][s * 32 + g * 16 + 8];
                s0 = wmma_bf16(qf[s], u0.v, s0);
                s1 = wmma_bf16(qf[s], u1.v, s1);
            }

            const float scale = 0.125f;  // 1/sqrt(64)
            int kcol0 = kb * 32 + lr;
            int kcol1 = kb * 32 + 16 + lr;

#pragma unroll
            for (int r = 0; r < 8; ++r) {
                int qrow = q0 + g * 8 + r;
                float v0 = s0[r] * scale;
                float v1 = s1[r] * scale;
                if (kcol0 > qrow) v0 = -3.0e38f;  // causal mask
                if (kcol1 > qrow) v1 = -3.0e38f;

                float mx = fmaxf(v0, v1);
#pragma unroll
                for (int d = 1; d < 16; d <<= 1) mx = fmaxf(mx, __shfl_xor(mx, d, 32));
                float mnew  = fmaxf(mrun[r], mx);
                float alpha = __expf(mrun[r] - mnew);
                float p0 = __expf(v0 - mnew);
                float p1 = __expf(v1 - mnew);
                float ps = p0 + p1;
#pragma unroll
                for (int d = 1; d < 16; d <<= 1) ps += __shfl_xor(ps, d, 32);
                lrun[r] = lrun[r] * alpha + ps;
                mrun[r] = mnew;
#pragma unroll
                for (int c = 0; c < 4; ++c) cacc[c][r] *= alpha;

                Pst[wave][g * 8 + r][lr]      = (bf16)p0;
                Pst[wave][g * 8 + r][16 + lr] = (bf16)p1;
            }

            // DS ops are in-order per wave; drain writes before the re-read.
            asm volatile("s_wait_dscnt 0x0" ::: "memory");

            V16U up;
            up.h[0] = *(const v8bf*)&Pst[wave][lr][g * 16];
            up.h[1] = *(const v8bf*)&Pst[wave][lr][g * 16 + 8];
            v16bf pf = up.v;

#pragma unroll
            for (int c = 0; c < 4; ++c) {
                V16U uv;
                uv.h[0] = *(const v8bf*)&Vt[c * 16 + lr][g * 16];
                uv.h[1] = *(const v8bf*)&Vt[c * 16 + lr][g * 16 + 8];
                cacc[c] = wmma_bf16(pf, uv.v, cacc[c]);
            }
        }
    }

#pragma unroll
    for (int c = 0; c < 4; ++c) {
#pragma unroll
        for (int r = 0; r < 8; ++r) {
            int qrow = q0 + g * 8 + r;
            float v  = cacc[c][r] / lrun[r];
            ctx[(size_t)(b * T + qrow) * (H * DK) + h * DK + c * 16 + lr] = (bf16)v;
        }
    }
}

// ---------------------------------------------------------------------------
// Host-side orchestration.
// ---------------------------------------------------------------------------
static inline size_t align256(size_t x) { return (x + 255) & ~(size_t)255; }

extern "C" void kernel_launch(void* const* d_in, const int* in_sizes, int n_in,
                              void* d_out, int out_size, void* d_ws, size_t ws_size,
                              hipStream_t stream) {
    (void)in_sizes; (void)n_in; (void)out_size; (void)ws_size;
    const int B = 2, T = 2048, D = 1024, Hh = 16, DK = 64;
    const int M  = B * T;        // 4096 tokens
    const int N3 = 3 * Hh * DK;  // 3072
    const int DF = 4 * D;        // 4096

    const float* X     = (const float*)d_in[0];
    const float* Wqkv  = (const float*)d_in[1];
    const float* bqkv  = (const float*)d_in[2];
    const float* Wo    = (const float*)d_in[3];
    const float* bo    = (const float*)d_in[4];
    const float* W1    = (const float*)d_in[5];
    const float* b1    = (const float*)d_in[6];
    const float* W2    = (const float*)d_in[7];
    const float* b2    = (const float*)d_in[8];
    const float* ln1_g = (const float*)d_in[9];
    const float* ln1_b = (const float*)d_in[10];
    const float* ln2_g = (const float*)d_in[11];
    const float* ln2_b = (const float*)d_in[12];
    float* out = (float*)d_out;

    // Workspace carve (bf16 unless noted), ~126 MB total.
    char* ws = (char*)d_ws;
    size_t off = 0;
    bf16* WqkvT = (bf16*)(ws + off); off += align256((size_t)D * N3 * 2);  // [3072][1024]
    bf16* WoT   = (bf16*)(ws + off); off += align256((size_t)D * D * 2);   // [1024][1024]
    bf16* W1T   = (bf16*)(ws + off); off += align256((size_t)D * DF * 2);  // [4096][1024]
    bf16* W2T   = (bf16*)(ws + off); off += align256((size_t)DF * D * 2);  // [1024][4096]
    bf16* Xn_b  = (bf16*)(ws + off); off += align256((size_t)M * D * 2);
    bf16* qkv_b = (bf16*)(ws + off); off += align256((size_t)M * N3 * 2);
    bf16* ctx_b = (bf16*)(ws + off); off += align256((size_t)M * D * 2);
    float* X1_f = (float*)(ws + off); off += align256((size_t)M * D * 4);
    bf16* Xn2_b = (bf16*)(ws + off); off += align256((size_t)M * D * 2);
    bf16* H_b   = (bf16*)(ws + off); off += align256((size_t)M * DF * 2);

    // 1) Transpose-convert weights to bf16 [N][K] so GEMM B-staging is
    //    a contiguous async copy.
    cvt_t_kernel<<<dim3(D / 32, N3 / 32), 256, 0, stream>>>(Wqkv, WqkvT, D, N3);
    cvt_t_kernel<<<dim3(D / 32, D / 32), 256, 0, stream>>>(Wo, WoT, D, D);
    cvt_t_kernel<<<dim3(D / 32, DF / 32), 256, 0, stream>>>(W1, W1T, D, DF);
    cvt_t_kernel<<<dim3(DF / 32, D / 32), 256, 0, stream>>>(W2, W2T, DF, D);

    // 2) LN1 -> bf16
    ln_bf16_kernel<<<M, 256, 0, stream>>>(X, ln1_g, ln1_b, Xn_b, D);

    // 3) QKV GEMM: [4096,1024] @ [1024,3072] -> bf16
    gemm_bf16_kernel<<<dim3(M / GM, N3 / GN), 256, 0, stream>>>(
        Xn_b, WqkvT, bqkv, nullptr, nullptr, qkv_b, M, N3, D, 0);

    // 4) Causal flash attention -> ctx bf16 [4096,1024]
    attn_kernel<<<dim3(T / 128, B * Hh), 256, 0, stream>>>(qkv_b, ctx_b, B, T);

    // 5) Output proj + residual: X1 = X + ctx @ Wo + bo  (fp32)
    gemm_bf16_kernel<<<dim3(M / GM, D / GN), 256, 0, stream>>>(
        ctx_b, WoT, bo, X, X1_f, nullptr, M, D, D, 0);

    // 6) LN2 -> bf16
    ln_bf16_kernel<<<M, 256, 0, stream>>>(X1_f, ln2_g, ln2_b, Xn2_b, D);

    // 7) FFN1 + ReLU: H = relu(Xn2 @ W1 + b1) -> bf16 [4096,4096]
    gemm_bf16_kernel<<<dim3(M / GM, DF / GN), 256, 0, stream>>>(
        Xn2_b, W1T, b1, nullptr, nullptr, H_b, M, DF, D, 1);

    // 8) FFN2 + residual: out = X1 + H @ W2 + b2 (fp32)
    gemm_bf16_kernel<<<dim3(M / GM, D / GN), 256, 0, stream>>>(
        H_b, W2T, b2, X1_f, out, nullptr, M, D, DF, 0);
}
